// LSTMPredictor_73083163509204
// MI455X (gfx1250) — matrix operand order repaired and checked by reading, not compiled, not run on hardware
//
#include <hip/hip_runtime.h>
#include <math.h>

// ---------------------------------------------------------------------------
// 2-layer LSTM, B=128, T=512, H=512 — persistent WMMA kernel for gfx1250.
// Weights live in LDS (192 rows x 1040B = 195KB of the WGP's 320KB),
// preloaded once per kernel with global_load_async_to_lds_b128 (ASYNCcnt),
// then reused for all 512 timesteps. Only h-state traffic touches L2 per step.
// ---------------------------------------------------------------------------

namespace {
constexpr int Bn = 128;           // batch
constexpr int Tn = 512;           // sequence length
constexpr int Hn = 512;           // hidden
constexpr int Gn = 4 * Hn;        // gate width (i,f,g,o)
constexpr int NBLK = 32;          // one block per h-tile
constexpr int NTHR = 256;         // 8 wave32 -> 8 row-tiles
constexpr unsigned LDS_ROW = 1040;       // bytes per weight row (512 f16 + pad)
constexpr unsigned LDS_BYTES = 192 * LDS_ROW;  // 3 matrices x 64 rows
}

typedef _Float16 v16h __attribute__((ext_vector_type(16)));
typedef _Float16 v8h  __attribute__((ext_vector_type(8)));
typedef float    v8f  __attribute__((ext_vector_type(8)));

using lds_v16h = __attribute__((address_space(3))) const v16h;

struct WsPtrs {
  unsigned* bar;          // [2] grid-barrier {arrive, generation}
  _Float16* Whh1;         // (4H, H) f16 row-major
  _Float16* Wih2;         // (4H, H) f16 row-major
  _Float16* Whh2;         // (4H, H) f16 row-major
  float*    bias1;        // (4H) b_ih1 + b_hh1
  float*    bias2;        // (4H) b_ih2 + b_hh2
  float*    c1;           // (B, H) f32
  float*    c2;           // (B, H) f32
  _Float16* h1[2];        // (B, H) f16 ping-pong
  _Float16* h2[2];        // (B, H) f16 ping-pong
};

// ---------------------------------------------------------------------------
// helpers
// ---------------------------------------------------------------------------

__device__ __forceinline__ float sigm(float x) {
  return 1.0f / (1.0f + __expf(-x));
}
__device__ __forceinline__ float logsig(float x) {
  return fminf(x, 0.0f) - log1pf(__expf(-fabsf(x)));
}

__device__ __forceinline__ v8f wmma_f16(v16h a, v16h b, v8f c) {
  return __builtin_amdgcn_wmma_f32_16x16x32_f16(
      /*neg_a=*/false, a, /*neg_b=*/false, b,
      /*c_mod=*/(short)0, c, /*reuse_a=*/false, /*reuse_b=*/false);
}

// A fragment: 16x32 (MxK) f16, lane l holds M = l&15.
//   lanes 0-15 : elems 0..15 -> K=kb+0..15
//   lanes 16-31: elems 0..15 -> K=kb+16..31
__device__ __forceinline__ v16h load_a_frag(const _Float16* h, int row, int kb,
                                            int lane) {
  const _Float16* p = h + row * Hn + kb + ((lane >> 4) & 1) * 16;
  v8h lo = *reinterpret_cast<const v8h*>(p);
  v8h hi = *reinterpret_cast<const v8h*>(p + 8);
  v16h a;
#pragma unroll
  for (int i = 0; i < 8; ++i) {
    a[i]     = lo[i];
    a[i + 8] = hi[i];
  }
  return a;
}

// B fragment read from LDS. Weight slice layout: 192 rows (3 matrices x 4
// gates x 16 cols), row stride LDS_ROW bytes, K contiguous f16 within a row.
// Lane l: N-col = l&15, K base offset = kb + (l>=16 ? 16 : 0)  (32B read).
__device__ __forceinline__ v16h lds_b_frag(unsigned rowByteOff, int kb) {
  unsigned off = rowByteOff + (unsigned)(kb * 2);
  lds_v16h* p = (lds_v16h*)off;   // addrspace(3): raw LDS byte offset
  return *p;
}

__device__ __forceinline__ void grid_barrier(unsigned* bar) {
  __syncthreads();
  if (threadIdx.x == 0) {
    __threadfence();
    unsigned gen =
        __hip_atomic_load(&bar[1], __ATOMIC_ACQUIRE, __HIP_MEMORY_SCOPE_AGENT);
    unsigned prev = __hip_atomic_fetch_add(&bar[0], 1u, __ATOMIC_ACQ_REL,
                                           __HIP_MEMORY_SCOPE_AGENT);
    if (prev == (unsigned)(NBLK - 1)) {
      __hip_atomic_store(&bar[0], 0u, __ATOMIC_RELAXED,
                         __HIP_MEMORY_SCOPE_AGENT);
      __hip_atomic_fetch_add(&bar[1], 1u, __ATOMIC_ACQ_REL,
                             __HIP_MEMORY_SCOPE_AGENT);
    } else {
      while (__hip_atomic_load(&bar[1], __ATOMIC_ACQUIRE,
                               __HIP_MEMORY_SCOPE_AGENT) == gen) {
        __builtin_amdgcn_s_sleep(2);
      }
    }
    __threadfence();
  }
  __syncthreads();
}

// ---------------------------------------------------------------------------
// init: f32->f16 weight copies, fused biases, zero states, reset barrier
// ---------------------------------------------------------------------------

__global__ void lstm_init(const float* Whh1, const float* Wih2,
                          const float* Whh2, const float* bih1,
                          const float* bhh1, const float* bih2,
                          const float* bhh2, WsPtrs ws) {
  int i = blockIdx.x * blockDim.x + threadIdx.x;
  int stride = gridDim.x * blockDim.x;
  for (int k = i; k < Gn * Hn; k += stride) {
    ws.Whh1[k] = (_Float16)Whh1[k];
    ws.Wih2[k] = (_Float16)Wih2[k];
    ws.Whh2[k] = (_Float16)Whh2[k];
  }
  for (int k = i; k < Gn; k += stride) {
    ws.bias1[k] = bih1[k] + bhh1[k];
    ws.bias2[k] = bih2[k] + bhh2[k];
  }
  for (int k = i; k < Bn * Hn; k += stride) {
    ws.c1[k] = 0.0f;
    ws.c2[k] = 0.0f;
    ws.h1[0][k] = (_Float16)0.0f;
    ws.h2[0][k] = (_Float16)0.0f;
  }
  if (i == 0) {
    ws.bar[0] = 0u;
    ws.bar[1] = 0u;
  }
}

// ---------------------------------------------------------------------------
// persistent LSTM kernel.
// blockIdx.x = h-tile (0..31); wave = row-tile (0..7). A block's weight slice
// (its 16 h-columns of all 4 gates of all 3 matrices) is async-preloaded to
// LDS once, then all 512 timesteps read B fragments from LDS.
// ---------------------------------------------------------------------------

__global__ __launch_bounds__(NTHR) void lstm_persistent(
    const float* __restrict__ y, const float* __restrict__ Wih1, WsPtrs ws,
    float* __restrict__ out) {
  const int lane = threadIdx.x & 31;
  const int wave = threadIdx.x >> 5;
  const int ht = blockIdx.x;                 // 0..31 h tile
  const int r = wave;                        // 0..7  row tile
  const int rbase = r * 16;
  const int ncol = ht * 16 + (lane & 15);    // h column (B-frag N / C col)
  const int arow = rbase + (lane & 15);      // A-frag row
  const int hrow = ((lane >> 4) & 1) * 8;    // C row offset of this lane

  const size_t BTH = (size_t)Bn * Tn * Hn;

  // ---- one-time async weight preload: 192 rows x 1024B, 16B chunks ----
  {
    const _Float16* wsrc[3] = {ws.Whh1, ws.Wih2, ws.Whh2};
    // 192 rows * 64 chunks = 12288 chunks; 48 per thread, uniform trip count.
    for (int ch = threadIdx.x; ch < 192 * 64; ch += NTHR) {
      const int row = ch >> 6;               // 0..191
      const int m = row >> 6;                // matrix 0..2
      const int lr = row & 63;               // gate*16 + col
      const int g = lr >> 4;
      const int c = lr & 15;
      const int n = g * Hn + ht * 16 + c;    // global weight row
      const _Float16* gp = wsrc[m] + (size_t)n * Hn + (ch & 63) * 8;
      const unsigned ldsoff = (unsigned)row * LDS_ROW + (unsigned)(ch & 63) * 16;
      asm volatile("global_load_async_to_lds_b128 %0, %1, off"
                   :: "v"(ldsoff), "v"(gp)
                   : "memory");
    }
    asm volatile("s_wait_asynccnt 0x0" ::: "memory");
    __syncthreads();
  }

  // per-lane LDS row byte offsets for the 12 B-fragment streams
  // rowByteOff(m, g) = (m*64 + g*16 + (lane&15)) * LDS_ROW + koff*2
  unsigned bofs[3][4];
#pragma unroll
  for (int m = 0; m < 3; ++m)
#pragma unroll
    for (int g = 0; g < 4; ++g)
      bofs[m][g] = (unsigned)((m * 64 + g * 16 + (lane & 15)) * LDS_ROW) +
                   (unsigned)(((lane >> 4) & 1) * 32);

  for (int t = 0; t < Tn; ++t) {
    const _Float16* h1r = ws.h1[t & 1];
    _Float16*       h1w = ws.h1[(t + 1) & 1];
    const _Float16* h2r = ws.h2[t & 1];
    _Float16*       h2w = ws.h2[(t + 1) & 1];

    // ------------------------- layer 1 -------------------------
    v8f ai, af, ag, ao;
    {
      const float bi = ws.bias1[0 * Hn + ncol];
      const float bf = ws.bias1[1 * Hn + ncol];
      const float bg = ws.bias1[2 * Hn + ncol];
      const float bo = ws.bias1[3 * Hn + ncol];
      const float wi = Wih1[0 * Hn + ncol];  // W_ih1 is (4H, 1)
      const float wf = Wih1[1 * Hn + ncol];
      const float wg = Wih1[2 * Hn + ncol];
      const float wo = Wih1[3 * Hn + ncol];
#pragma unroll
      for (int e = 0; e < 8; ++e) {
        const int b = rbase + hrow + e;
        const float x = y[(size_t)b * Tn + t];  // rank-1 input term
        ai[e] = bi + x * wi;
        af[e] = bf + x * wf;
        ag[e] = bg + x * wg;
        ao[e] = bo + x * wo;
      }
    }
#pragma unroll 4
    for (int kb = 0; kb < Hn; kb += 32) {
      const v16h a = load_a_frag(h1r, arow, kb, lane);
      ai = wmma_f16(a, lds_b_frag(bofs[0][0], kb), ai);
      af = wmma_f16(a, lds_b_frag(bofs[0][1], kb), af);
      ag = wmma_f16(a, lds_b_frag(bofs[0][2], kb), ag);
      ao = wmma_f16(a, lds_b_frag(bofs[0][3], kb), ao);
    }
#pragma unroll
    for (int e = 0; e < 8; ++e) {
      const int b = rbase + hrow + e;
      const int idx = b * Hn + ncol;
      const float c = ws.c1[idx];
      const float ig = sigm(ai[e]);
      const float fg = sigm(af[e]);
      const float gg = tanhf(ag[e]);
      const float og = sigm(ao[e]);
      const float cn = fg * c + ig * gg;
      ws.c1[idx] = cn;
      h1w[idx] = (_Float16)(og * tanhf(cn));
    }

    grid_barrier(ws.bar);  // h1(t) complete before layer-2 GEMM reads it

    // ------------------------- layer 2 -------------------------
    {
      const float bi = ws.bias2[0 * Hn + ncol];
      const float bf = ws.bias2[1 * Hn + ncol];
      const float bg = ws.bias2[2 * Hn + ncol];
      const float bo = ws.bias2[3 * Hn + ncol];
#pragma unroll
      for (int e = 0; e < 8; ++e) {
        ai[e] = bi;
        af[e] = bf;
        ag[e] = bg;
        ao[e] = bo;
      }
    }
#pragma unroll 2
    for (int kb = 0; kb < Hn; kb += 32) {
      const v16h a1 = load_a_frag(h1w, arow, kb, lane);  // input = h1(t)
      ai = wmma_f16(a1, lds_b_frag(bofs[1][0], kb), ai);
      af = wmma_f16(a1, lds_b_frag(bofs[1][1], kb), af);
      ag = wmma_f16(a1, lds_b_frag(bofs[1][2], kb), ag);
      ao = wmma_f16(a1, lds_b_frag(bofs[1][3], kb), ao);
      const v16h a2 = load_a_frag(h2r, arow, kb, lane);  // recurrent h2(t-1)
      ai = wmma_f16(a2, lds_b_frag(bofs[2][0], kb), ai);
      af = wmma_f16(a2, lds_b_frag(bofs[2][1], kb), af);
      ag = wmma_f16(a2, lds_b_frag(bofs[2][2], kb), ag);
      ao = wmma_f16(a2, lds_b_frag(bofs[2][3], kb), ao);
    }
#pragma unroll
    for (int e = 0; e < 8; ++e) {
      const int b = rbase + hrow + e;
      const int idx = b * Hn + ncol;
      const float c = ws.c2[idx];
      const float ig = sigm(ai[e]);
      const float fg = sigm(af[e]);
      const float gg = tanhf(ag[e]);
      const float og = sigm(ao[e]);
      const float cn = fg * c + ig * gg;
      ws.c2[idx] = cn;
      const float hn = og * tanhf(cn);
      h2w[idx] = (_Float16)hn;
      const float ls = logsig(hn);
      const size_t oidx = (size_t)b * Tn * Hn + (size_t)t * Hn + ncol;
      out[oidx] = ls;        // output1
      out[BTH + oidx] = ls;  // output2 (reference returns the tensor twice)
    }

    grid_barrier(ws.bar);  // h2(t) complete before next step reads it
  }
}

// ---------------------------------------------------------------------------
// host entry
// ---------------------------------------------------------------------------

extern "C" void kernel_launch(void* const* d_in, const int* in_sizes, int n_in,
                              void* d_out, int out_size, void* d_ws,
                              size_t ws_size, hipStream_t stream) {
  const float* y    = (const float*)d_in[0];
  const float* Wih1 = (const float*)d_in[1];
  const float* Whh1 = (const float*)d_in[2];
  const float* bih1 = (const float*)d_in[3];
  const float* bhh1 = (const float*)d_in[4];
  const float* Wih2 = (const float*)d_in[5];
  const float* Whh2 = (const float*)d_in[6];
  const float* bih2 = (const float*)d_in[7];
  const float* bhh2 = (const float*)d_in[8];
  (void)in_sizes; (void)n_in; (void)out_size; (void)ws_size;

  char* base = (char*)d_ws;
  size_t off = 0;
  auto carve = [&](size_t bytes) -> void* {
    off = (off + 255) & ~(size_t)255;
    void* p = base + off;
    off += bytes;
    return p;
  };

  WsPtrs ws;
  ws.bar   = (unsigned*)carve(2 * sizeof(unsigned));
  ws.Whh1  = (_Float16*)carve((size_t)Gn * Hn * sizeof(_Float16));
  ws.Wih2  = (_Float16*)carve((size_t)Gn * Hn * sizeof(_Float16));
  ws.Whh2  = (_Float16*)carve((size_t)Gn * Hn * sizeof(_Float16));
  ws.bias1 = (float*)carve((size_t)Gn * sizeof(float));
  ws.bias2 = (float*)carve((size_t)Gn * sizeof(float));
  ws.c1    = (float*)carve((size_t)Bn * Hn * sizeof(float));
  ws.c2    = (float*)carve((size_t)Bn * Hn * sizeof(float));
  ws.h1[0] = (_Float16*)carve((size_t)Bn * Hn * sizeof(_Float16));
  ws.h1[1] = (_Float16*)carve((size_t)Bn * Hn * sizeof(_Float16));
  ws.h2[0] = (_Float16*)carve((size_t)Bn * Hn * sizeof(_Float16));
  ws.h2[1] = (_Float16*)carve((size_t)Bn * Hn * sizeof(_Float16));

  lstm_init<<<1024, 256, 0, stream>>>(Whh1, Wih2, Whh2, bih1, bhh1, bih2, bhh2,
                                      ws);
  lstm_persistent<<<NBLK, NTHR, LDS_BYTES, stream>>>(y, Wih1, ws,
                                                     (float*)d_out);
}
